// AttentiveDecisionTree_29807073034342
// MI455X (gfx1250) — matrix operand order repaired and checked by reading, not compiled, not run on hardware
//
#include <hip/hip_runtime.h>
#include <hip/hip_bf16.h>

typedef __attribute__((ext_vector_type(16))) _Float16 v16h;
typedef __attribute__((ext_vector_type(8)))  float    v8f;

union Frag { v16h v; float4 f4[2]; _Float16 h[16]; };

#define B_TOT   4096
#define F_DIM   256
#define M_MEM   32
#define S_SEQ   33      // 1 + M
#define HK      8       // H*K
#define NCOL_Q  8
#define NCOL_K  264     // 33*8
#define NCOL_V  264
#define NCOL    536
#define NCOL_P  544     // padded to 34*16
#define NTREE   256
#define DEPTH   6
#define NJ      1536    // NTREE*DEPTH
#define LLEAF   64
#define UNITS   16

// ---------------------------------------------------------------------------
// Kernel 1: sparsemax over F=256 per (tree,depth) row via bisection on tau.
// g(tau) = sum(max(z - tau, 0)) is continuous & decreasing; bracket
// [zmax-1, zmax] always contains the root of g(tau)=1. 40 bisection steps
// reach ~1e-12 of the bracket. Output stored as f16 (WMMA B operand).
// ---------------------------------------------------------------------------
__global__ __launch_bounds__(256) void k_sparsemax(const float* __restrict__ fsl,
                                                   _Float16* __restrict__ fs_h) {
    const int j   = blockIdx.x;        // 0..1535
    const int tid = threadIdx.x;       // 0..255 (one feature each)
    const float z = fsl[j * 256 + tid];

    __shared__ float wred[8];

    // block max
    float m = z;
#pragma unroll
    for (int o = 16; o > 0; o >>= 1) m = fmaxf(m, __shfl_xor(m, o, 32));
    if ((tid & 31) == 0) wred[tid >> 5] = m;
    __syncthreads();
    float zmax = wred[0];
#pragma unroll
    for (int i = 1; i < 8; ++i) zmax = fmaxf(zmax, wred[i]);

    float lo = zmax - 1.0f, hi = zmax;
    for (int it = 0; it < 40; ++it) {
        float mid = 0.5f * (lo + hi);
        float v = fmaxf(z - mid, 0.0f);
#pragma unroll
        for (int o = 16; o > 0; o >>= 1) v += __shfl_xor(v, o, 32);
        __syncthreads();                     // protect wred reuse
        if ((tid & 31) == 0) wred[tid >> 5] = v;
        __syncthreads();
        float s = 0.0f;
#pragma unroll
        for (int i = 0; i < 8; ++i) s += wred[i];
        if (s >= 1.0f) lo = mid; else hi = mid;   // uniform branch
    }
    float tau = 0.5f * (lo + hi);
    fs_h[j * 256 + tid] = (_Float16)fmaxf(z - tau, 0.0f);
}

// ---------------------------------------------------------------------------
// Kernel 2: build the fused attention weight matrix Wbig[col][f] (f16,
// column-major, 544 cols padded) + threshold / exp(-log_temp) tables.
//   cols 0..7     : q   -> Wq[f, hk]
//   cols 8..271   : k_t -> mem_t[f] * Wk[f, hk]   (mem_0 = ones)
//   cols 272..535 : v_t -> mem_t[f] * Wv[f, hk]
//   cols 536..543 : zero pad
// ---------------------------------------------------------------------------
__global__ __launch_bounds__(256) void k_prep(const float* __restrict__ memory,
                                              const float* __restrict__ Wq,
                                              const float* __restrict__ Wk,
                                              const float* __restrict__ Wv,
                                              const float* __restrict__ thresholds,
                                              const float* __restrict__ log_temp,
                                              _Float16* __restrict__ wbig,
                                              float* __restrict__ thrv,
                                              float* __restrict__ expv) {
    const int b = blockIdx.x, tid = threadIdx.x;
    if (b < NCOL_P) {
        const int f = tid, col = b;
        float w;
        if (col < 8) {
            w = Wq[f * 8 + col];
        } else if (col < 272) {
            int t = (col - 8) >> 3, hk = (col - 8) & 7;
            float mf = (t == 0) ? 1.0f : memory[(t - 1) * 256 + f];
            w = mf * Wk[f * 8 + hk];
        } else if (col < 536) {
            int t = (col - 272) >> 3, hk = (col - 272) & 7;
            float mf = (t == 0) ? 1.0f : memory[(t - 1) * 256 + f];
            w = mf * Wv[f * 8 + hk];
        } else {
            w = 0.0f;
        }
        wbig[col * 256 + f] = (_Float16)w;
    } else {
        int j = (b - NCOL_P) * 256 + tid;
        if (j < NJ) { thrv[j] = thresholds[j]; expv[j] = expf(-log_temp[j]); }
    }
}

// ---------------------------------------------------------------------------
// Fragment loaders per CDNA5 ISA VGPR layouts (wave32, 16x16x32 f16 WMMA).
//   A (16x32):  lane m = lane&15; halfs 0..7  -> K = hs+0..7,
//                                 halfs 8..15 -> K = 16+hs+0..7,  hs = (lane>>4)*8
//   B (32x16):  lane col = lane&15; halfs 0..7 -> K = hs+0..7, halfs 8..15 -> K = 16+hs+0..7
//   C/D (16x16 f32): vgpr i -> row i + 8*(lane>>4), col = lane&15
// ---------------------------------------------------------------------------
__device__ __forceinline__ void load_a_frags(Frag afr[8],
                                             const _Float16 (*xh)[256], int lane) {
    const int m  = lane & 15;
    const int hs = (lane >> 4) * 8;
#pragma unroll
    for (int s = 0; s < 8; ++s) {
        afr[s].f4[0] = *(const float4*)&xh[m][s * 32 + hs];
        afr[s].f4[1] = *(const float4*)&xh[m][s * 32 + 16 + hs];
    }
}

__device__ __forceinline__ Frag load_b_frag(const _Float16* __restrict__ base,
                                            int ntile, int kstep, int lane) {
    Frag bf;
    const _Float16* p = base + (size_t)(ntile * 16 + (lane & 15)) * 256
                             + kstep * 32 + ((lane >> 4) << 3);
    bf.f4[0] = *(const float4*)(p);
    bf.f4[1] = *(const float4*)(p + 16);
    return bf;
}

// ---------------------------------------------------------------------------
// Kernel 3: fused attention + ODT for one 16-row batch tile.
// ---------------------------------------------------------------------------
__global__ __launch_bounds__(256) void k_main(const float* __restrict__ inputs,
                                              const float* __restrict__ bq,
                                              const float* __restrict__ bk,
                                              const float* __restrict__ bv,
                                              const float* __restrict__ Wo,
                                              const float* __restrict__ bo,
                                              const float* __restrict__ response,
                                              const _Float16* __restrict__ wbig,
                                              const _Float16* __restrict__ fs_h,
                                              const float* __restrict__ thrv,
                                              const float* __restrict__ expv,
                                              float* __restrict__ out) {
    __shared__ __align__(16) union {
        struct { float in32[16][256]; float P[16][NCOL_P]; } a;  // phases 0-3
        _Float16 bins[16][NJ];                                   // phases 4-6
    } u;
    __shared__ __align__(16) _Float16 xh[16][256];   // A tile: inputs f16, then x f16
    __shared__ float o_s[16][HK];
    __shared__ float outacc[16][UNITS];

    const int tid  = threadIdx.x;
    const int lane = tid & 31;
    const int wave = tid >> 5;
    const int bt   = blockIdx.x;                     // batch tile (16 rows)

    // ---- Phase 0: stage inputs tile (f32 for residual, f16 for WMMA A) ----
    for (int e = tid; e < 16 * 256; e += 256) {
        int r = e >> 8, f = e & 255;
        float v = inputs[(size_t)(bt * 16 + r) * 256 + f];
        u.a.in32[r][f] = v;
        xh[r][f] = (_Float16)v;
    }
    __syncthreads();

    // ---- Phase 1: P = X @ Wbig  (16x256 @ 256x544), f16 WMMA, f32 acc ----
    {
        Frag afr[8];
        load_a_frags(afr, xh, lane);
        for (int nt = wave; nt < 34; nt += 8) {
            v8f acc = {0.f, 0.f, 0.f, 0.f, 0.f, 0.f, 0.f, 0.f};
#pragma unroll
            for (int s = 0; s < 8; ++s) {
                Frag bf = load_b_frag(wbig, nt, s, lane);
                acc = __builtin_amdgcn_wmma_f32_16x16x32_f16(
                        false, afr[s].v, false, bf.v, (short)0, acc, false, false);
            }
            const int col = nt * 16 + (lane & 15);
            const int rb  = (lane >> 4) * 8;
#pragma unroll
            for (int i = 0; i < 8; ++i) u.a.P[rb + i][col] = acc[i];
        }
    }
    __syncthreads();

    // ---- Phase 2: per (row, head) softmax over 33 positions ----
    if (tid < 64) {
        const int r = tid >> 2, h = tid & 3;
        const float bq0 = bq[h * 2], bq1 = bq[h * 2 + 1];
        const float bk0 = bk[h * 2], bk1 = bk[h * 2 + 1];
        const float bv0 = bv[h * 2], bv1 = bv[h * 2 + 1];
        const float q0 = u.a.P[r][h * 2] + bq0;
        const float q1 = u.a.P[r][h * 2 + 1] + bq1;
        const float inv_sqrt_k = 0.70710678118654752f;   // 1/sqrt(K=2)
        float sc[S_SEQ], mx = -1e30f;
#pragma unroll
        for (int t = 0; t < S_SEQ; ++t) {
            float k0 = u.a.P[r][8 + t * 8 + h * 2] + bk0;
            float k1 = u.a.P[r][8 + t * 8 + h * 2 + 1] + bk1;
            float s = (q0 * k0 + q1 * k1) * inv_sqrt_k;
            sc[t] = s; mx = fmaxf(mx, s);
        }
        float den = 0.0f;
#pragma unroll
        for (int t = 0; t < S_SEQ; ++t) { float e = expf(sc[t] - mx); sc[t] = e; den += e; }
        const float rden = 1.0f / den;
        float o0 = 0.0f, o1 = 0.0f;
#pragma unroll
        for (int t = 0; t < S_SEQ; ++t) {
            float a = sc[t] * rden;
            o0 += a * (u.a.P[r][272 + t * 8 + h * 2] + bv0);
            o1 += a * (u.a.P[r][272 + t * 8 + h * 2 + 1] + bv1);
        }
        o_s[r][h * 2] = o0; o_s[r][h * 2 + 1] = o1;
    }
    __syncthreads();

    // ---- Phase 3: x = inputs + o @ Wo + bo, restage as f16 A tile ----
    {
        const int f = tid;
        const float bof = bo[f];
        float wv[HK];
#pragma unroll
        for (int hk = 0; hk < HK; ++hk) wv[hk] = Wo[hk * 256 + f];
        for (int r = 0; r < 16; ++r) {
            float acc = u.a.in32[r][f] + bof;
#pragma unroll
            for (int hk = 0; hk < HK; ++hk) acc += o_s[r][hk] * wv[hk];
            xh[r][f] = (_Float16)acc;
        }
    }
    __syncthreads();

    // ---- Phase 4: feat = X @ fs^T (16x256 @ 256x1536) via WMMA, with
    //      inline threshold/temp/clip -> bins (f16) straight from the
    //      accumulator (saves a 96KB f32 stage + a full pass). ----
    {
        Frag afr[8];
        load_a_frags(afr, xh, lane);
        for (int nt = wave; nt < 96; nt += 8) {
            v8f acc = {0.f, 0.f, 0.f, 0.f, 0.f, 0.f, 0.f, 0.f};
#pragma unroll
            for (int s = 0; s < 8; ++s) {
                Frag bf = load_b_frag(fs_h, nt, s, lane);
                acc = __builtin_amdgcn_wmma_f32_16x16x32_f16(
                        false, afr[s].v, false, bf.v, (short)0, acc, false, false);
            }
            const int col = nt * 16 + (lane & 15);    // j = tree*6 + depth
            const int rb  = (lane >> 4) * 8;
            const float th = thrv[col];
            const float ex = expv[col];
#pragma unroll
            for (int i = 0; i < 8; ++i) {
                float lg = (acc[i] - th) * ex;
                float bn = fminf(fmaxf(0.5f * lg + 0.5f, 0.0f), 1.0f);
                u.bins[rb + i][col] = (_Float16)bn;
            }
        }
    }
    __syncthreads();

    // ---- Phase 5: trees -> leaves -> response.  thread = (row b, 16-tree chunk) ----
    outacc[tid >> 4][tid & 15] = 0.0f;
    __syncthreads();
    {
        const int brow = tid >> 4, ch = tid & 15;
        float accu[UNITS];
#pragma unroll
        for (int i = 0; i < UNITS; ++i) accu[i] = 0.0f;
        for (int n = ch * 16; n < ch * 16 + 16; ++n) {
            float bn[DEPTH], cn[DEPTH];
#pragma unroll
            for (int d = 0; d < DEPTH; ++d) {
                float x = (float)u.bins[brow][n * DEPTH + d];
                bn[d] = x; cn[d] = 1.0f - x;
            }
            const float* rp = response + (size_t)n * LLEAF * UNITS;
            for (int l = 0; l < LLEAF; ++l) {
                float leaf = ((l & 1)  ? bn[0] : cn[0]) * ((l & 2)  ? bn[1] : cn[1])
                           * ((l & 4)  ? bn[2] : cn[2]) * ((l & 8)  ? bn[3] : cn[3])
                           * ((l & 16) ? bn[4] : cn[4]) * ((l & 32) ? bn[5] : cn[5]);
                const float4* r4 = (const float4*)(rp + l * UNITS);
                float4 a0 = r4[0], a1 = r4[1], a2 = r4[2], a3 = r4[3];
                accu[0]  += leaf * a0.x; accu[1]  += leaf * a0.y;
                accu[2]  += leaf * a0.z; accu[3]  += leaf * a0.w;
                accu[4]  += leaf * a1.x; accu[5]  += leaf * a1.y;
                accu[6]  += leaf * a1.z; accu[7]  += leaf * a1.w;
                accu[8]  += leaf * a2.x; accu[9]  += leaf * a2.y;
                accu[10] += leaf * a2.z; accu[11] += leaf * a2.w;
                accu[12] += leaf * a3.x; accu[13] += leaf * a3.y;
                accu[14] += leaf * a3.z; accu[15] += leaf * a3.w;
            }
        }
#pragma unroll
        for (int i = 0; i < UNITS; ++i) atomicAdd(&outacc[brow][i], accu[i]);
    }
    __syncthreads();
    out[(size_t)(bt * 16 + (tid >> 4)) * UNITS + (tid & 15)] = outacc[tid >> 4][tid & 15];
}

// ---------------------------------------------------------------------------
extern "C" void kernel_launch(void* const* d_in, const int* in_sizes, int n_in,
                              void* d_out, int out_size, void* d_ws, size_t ws_size,
                              hipStream_t stream) {
    const float* inputs     = (const float*)d_in[0];
    const float* memory     = (const float*)d_in[1];
    const float* Wq         = (const float*)d_in[2];
    const float* bq         = (const float*)d_in[3];
    const float* Wk         = (const float*)d_in[4];
    const float* bk         = (const float*)d_in[5];
    const float* Wv         = (const float*)d_in[6];
    const float* bv         = (const float*)d_in[7];
    const float* Wo         = (const float*)d_in[8];
    const float* bo         = (const float*)d_in[9];
    const float* fs_logits  = (const float*)d_in[10];
    const float* thresholds = (const float*)d_in[11];
    const float* log_temp   = (const float*)d_in[12];
    const float* response   = (const float*)d_in[13];
    float* out = (float*)d_out;

    // workspace layout (all 16B aligned)
    char* ws = (char*)d_ws;
    _Float16* fs_h  = (_Float16*)(ws);                              // 1536*256*2 = 786432
    _Float16* wbig  = (_Float16*)(ws + 786432);                     // 544*256*2  = 278528
    float*    thrv  = (float*)(ws + 786432 + 278528);               // 1536*4
    float*    expv  = (float*)(ws + 786432 + 278528 + 6144);        // 1536*4

    k_sparsemax<<<NJ, 256, 0, stream>>>(fs_logits, fs_h);
    k_prep<<<NCOL_P + 6, 256, 0, stream>>>(memory, Wq, Wk, Wv, thresholds, log_temp,
                                           wbig, thrv, expv);
    k_main<<<B_TOT / 16, 256, 0, stream>>>(inputs, bq, bk, bv, Wo, bo, response,
                                           wbig, fs_h, thrv, expv, out);
}
